// Attention_8787503088269
// MI455X (gfx1250) — matrix operand order repaired and checked by reading, not compiled, not run on hardware
//
#include <hip/hip_runtime.h>

// ---------------------------------------------------------------------------
// CDNA5 (gfx1250) attention block: qkv proj -> flash attention -> out proj.
// Matmuls: v_wmma_f32_16x16x32_bf16 (wave32, f32 accumulate).
// Data movement: global_load_async_to_lds_b128 (ASYNCcnt) for bf16 tiles,
// global_prefetch_b8 for GEMM streaming.
// ---------------------------------------------------------------------------

typedef __attribute__((ext_vector_type(16))) __bf16        v16bf;
typedef __attribute__((ext_vector_type(8)))  float         v8f;
typedef __attribute__((ext_vector_type(8)))  unsigned int  v8u;

// float -> bf16 with round-to-nearest-even (bit trick; NaN edge ignored)
__device__ __forceinline__ unsigned short f2bf(float f) {
    unsigned int u = __builtin_bit_cast(unsigned int, f);
    u += 0x7FFFu + ((u >> 16) & 1u);
    return (unsigned short)(u >> 16);
}

__device__ __forceinline__ v8f wmma_bf16(v16bf a, v16bf b, v8f c) {
    // (neg_a, A, neg_b, B, c_mod, C, reuse_a, reuse_b)
    return __builtin_amdgcn_wmma_f32_16x16x32_bf16(false, a, false, b,
                                                   (short)0, c, false, false);
}

// Async copy of 16 bytes/lane from global to LDS (GLOBAL_LOAD_ASYNC_TO_LDS_B128,
// tracked with ASYNCcnt). lds generic pointer's low 32 bits are the LDS address.
__device__ __forceinline__ void async_ld16(void* lds, const void* gptr) {
    unsigned int lds_off = (unsigned int)(unsigned long long)lds;
    asm volatile("global_load_async_to_lds_b128 %0, %1, off"
                 :: "v"(lds_off), "v"(gptr)
                 : "memory");
}
__device__ __forceinline__ void wait_async0() {
    asm volatile("s_wait_asynccnt 0x0" ::: "memory");
}

// A-fragment (16x32 bf16, M x K). LDS tile row-major, rowStrideU32 = row
// stride in dwords, kU32 = K start offset in dwords (pairs of bf16).
// Layout per ISA 7.12.2: lane L holds row M=L%16; lanes 0-15 get K 0-7/16-23,
// lanes 16-31 get K 8-15/24-31, two bf16 per VGPR.
__device__ __forceinline__ v16bf load_a_frag(const unsigned int* lds, int lane,
                                             int row0, int rowStrideU32, int kU32) {
    v8u r;
    const unsigned int* p = lds + (row0 + (lane & 15)) * rowStrideU32 + kU32;
    const int hi = (lane >> 4) * 4;
#pragma unroll
    for (int v = 0; v < 8; ++v)
        r[v] = p[(v >> 2) * 8 + hi + (v & 3)];
    return __builtin_bit_cast(v16bf, r);
}

// B-fragment (32x16 bf16, K x N). LDS tile holds B column-major:
// lds[n * rowStrideU32 + k/2]. Lane L holds column N=L%16; lanes 0-15 get
// K 0-7/16-23, lanes 16-31 get K 8-15/24-31, consecutive K pairs per VGPR.
__device__ __forceinline__ v16bf load_b_frag(const unsigned int* lds, int lane,
                                             int n0, int rowStrideU32, int kU32) {
    v8u r;
    const unsigned int* p =
        lds + (n0 + (lane & 15)) * rowStrideU32 + kU32;
    const int hi = (lane >> 4) * 4;
#pragma unroll
    for (int v = 0; v < 8; ++v)
        r[v] = p[(v >> 2) * 8 + hi + (v & 3)];
    return __builtin_bit_cast(v16bf, r);
}

// ---------------------------------------------------------------------------
// C[M,N] = A[M,K] @ B[K,N] + bias.  64x64 block tile, BK=32, 256 thr / 8 waves.
// A is f32 or bf16 (converted to bf16 in LDS); B is f32 (converted);
// C written as bf16 or f32.  M,N % 64 == 0, K % 32 == 0.
// ---------------------------------------------------------------------------
template <bool A_IS_BF16, bool OUT_BF16>
__global__ __launch_bounds__(256) void gemm_bias_kernel(
        const void* __restrict__ Aptr, const float* __restrict__ B,
        const float* __restrict__ bias, void* __restrict__ Cptr,
        int M, int N, int K) {
    __shared__ unsigned int AsU[64 * 16];   // 64 rows x 32 bf16 (row-major)
    __shared__ unsigned int BsU[64 * 16];   // B^T: 64 cols x 32 bf16
    unsigned short* As = (unsigned short*)AsU;
    unsigned short* Bs = (unsigned short*)BsU;

    const int tid  = threadIdx.x;
    const int lane = tid & 31;
    const int wave = tid >> 5;      // 0..7
    const int mt   = wave >> 1;     // row tile 0..3  -> rows 16*mt
    const int nh   = wave & 1;      // col half 0..1  -> cols 32*nh

    const int bm = blockIdx.y * 64;
    const int bn = blockIdx.x * 64;

    v8f acc0 = {}, acc1 = {};

    const int arow  = tid >> 2;         // 0..63
    const int akoff = (tid & 3) * 8;    // 0,8,16,24
    const int bk    = tid >> 3;         // 0..31
    const int bnn   = (tid & 7) * 8;    // 0..56

    for (int k0 = 0; k0 < K; k0 += 32) {
        // ---- stage A tile (convert to bf16 if needed) ----
        if constexpr (A_IS_BF16) {
            const unsigned short* A16 = (const unsigned short*)Aptr;
            const unsigned short* src = A16 + (size_t)(bm + arow) * K + k0 + akoff;
            async_ld16(As + arow * 32 + akoff, src);
            if (k0 + 32 < K) __builtin_prefetch(src + 32, 0, 3);
        } else {
            const float* A32 = (const float*)Aptr;
            const float4* src =
                (const float4*)(A32 + (size_t)(bm + arow) * K + k0 + akoff);
            float4 f0 = src[0], f1 = src[1];
            unsigned short* dst = As + arow * 32 + akoff;
            dst[0] = f2bf(f0.x); dst[1] = f2bf(f0.y);
            dst[2] = f2bf(f0.z); dst[3] = f2bf(f0.w);
            dst[4] = f2bf(f1.x); dst[5] = f2bf(f1.y);
            dst[6] = f2bf(f1.z); dst[7] = f2bf(f1.w);
            if (k0 + 32 < K) __builtin_prefetch((const float*)src + 32, 0, 3);
        }
        // ---- stage B tile transposed (column-major K) ----
        {
            const float* src = B + (size_t)(k0 + bk) * N + bn + bnn;
#pragma unroll
            for (int j = 0; j < 8; ++j)
                Bs[(bnn + j) * 32 + bk] = f2bf(src[j]);
            if (k0 + 32 < K) __builtin_prefetch(src + (size_t)32 * N, 0, 3);
        }
        if constexpr (A_IS_BF16) wait_async0();
        __syncthreads();

        v16bf a  = load_a_frag(AsU, lane, 16 * mt, 16, 0);
        v16bf b0 = load_b_frag(BsU, lane, 32 * nh, 16, 0);
        v16bf b1 = load_b_frag(BsU, lane, 32 * nh + 16, 16, 0);
        acc0 = wmma_bf16(a, b0, acc0);
        acc1 = wmma_bf16(a, b1, acc1);
        __syncthreads();
    }

    // ---- epilogue: bias add, store ----
    const int crow0 = bm + 16 * mt + (lane >> 4) * 8;
    const int col0  = bn + 32 * nh + (lane & 15);
    const float bias0 = bias[col0];
    const float bias1 = bias[col0 + 16];
#pragma unroll
    for (int r = 0; r < 8; ++r) {
        const size_t row = (size_t)(crow0 + r);
        const float v0 = acc0[r] + bias0;
        const float v1 = acc1[r] + bias1;
        if constexpr (OUT_BF16) {
            unsigned short* C = (unsigned short*)Cptr;
            C[row * N + col0]      = f2bf(v0);
            C[row * N + col0 + 16] = f2bf(v1);
        } else {
            float* C = (float*)Cptr;
            C[row * N + col0]      = v0;
            C[row * N + col0 + 16] = v1;
        }
    }
}

// ---------------------------------------------------------------------------
// Flash attention, causal. qkv: bf16 [B*T][2304] (q|k|v each 768 = 12 heads
// x 64). out: bf16 [B*T][768]. Block = (qb in 0..15, bh in 0..95), 128 thr /
// 4 waves; each wave owns 16 query rows of a 64-row q block.
// Q and K blocks staged with global_load_async_to_lds_b128; V staged
// transposed through VGPRs.
// ---------------------------------------------------------------------------
__global__ __launch_bounds__(128) void attn_kernel(
        const unsigned short* __restrict__ qkv,
        unsigned short* __restrict__ out) {
    constexpr int T = 1024, H = 768, QKV = 2304, HS = 64, NH = 12;

    __shared__ unsigned int QsU[64 * 32];   // [qr][d]  row-major
    __shared__ unsigned int KsU[64 * 32];   // [kr][d]  (== K^T column-major)
    __shared__ unsigned int VsU[64 * 32];   // V^T: [d][kr]
    __shared__ unsigned int PsU[64 * 32];   // [qr][kr] row-major
    unsigned short* Qs = (unsigned short*)QsU;
    unsigned short* Ks = (unsigned short*)KsU;
    unsigned short* Vs = (unsigned short*)VsU;
    unsigned short* Ps = (unsigned short*)PsU;

    const int tid  = threadIdx.x;
    const int lane = tid & 31;
    const int wave = tid >> 5;          // 0..3 -> q rows 16*wave..
    const int qb   = blockIdx.x;        // 0..15
    const int bh   = blockIdx.y;        // 0..95
    const int b    = bh / NH;
    const int h    = bh % NH;

    const size_t rowQ0 = (size_t)b * T + (size_t)qb * 64;
    const int rl0 = (lane >> 4) * 8;    // local row base (per C/D layout)
    const int cl  = lane & 15;          // local col

    // ---- load Q block (64x64 bf16) asynchronously ----
    {
        const int qr   = tid >> 1;
        const int doff = (tid & 1) * 32;
        const unsigned short* src = qkv + (rowQ0 + qr) * QKV + h * HS + doff;
        unsigned short* dst = Qs + qr * 64 + doff;
#pragma unroll
        for (int j = 0; j < 4; ++j)
            async_ld16(dst + j * 8, src + j * 8);
    }

    float m_i[8], l_i[8];
#pragma unroll
    for (int r = 0; r < 8; ++r) { m_i[r] = -1e30f; l_i[r] = 0.0f; }
    v8f oacc[4];
#pragma unroll
    for (int t = 0; t < 4; ++t) oacc[t] = (v8f){};

    for (int kb = 0; kb <= qb; ++kb) {
        __syncthreads();    // previous iter's frag reads done before overwrite
        // ---- load K block (async) and V block (transposed, through VGPRs) ----
        {
            const int kr   = tid >> 1;
            const int doff = (tid & 1) * 32;
            const size_t rowK = (size_t)b * T + (size_t)kb * 64 + kr;
            const unsigned short* ksrc = qkv + rowK * QKV + H + h * HS + doff;
            unsigned short* kdst = Ks + kr * 64 + doff;
#pragma unroll
            for (int j = 0; j < 4; ++j)
                async_ld16(kdst + j * 8, ksrc + j * 8);

            const uint4* vsrc =
                (const uint4*)(qkv + rowK * QKV + 2 * H + h * HS + doff);
            uint4 tmp[4];
#pragma unroll
            for (int j = 0; j < 4; ++j) tmp[j] = vsrc[j];
            const unsigned short* tv = (const unsigned short*)tmp;
#pragma unroll
            for (int d = 0; d < 32; ++d)
                Vs[(doff + d) * 64 + kr] = tv[d];
        }
        wait_async0();      // covers Q (first iter) + this K block
        __syncthreads();

        // ---- S = Q @ K^T  (16x64 per wave, 8 WMMAs) ----
        v8f sacc[4];
#pragma unroll
        for (int t = 0; t < 4; ++t) sacc[t] = (v8f){};
#pragma unroll
        for (int ks = 0; ks < 2; ++ks) {
            v16bf a = load_a_frag(QsU, lane, 16 * wave, 32, 16 * ks);
#pragma unroll
            for (int t = 0; t < 4; ++t) {
                v16bf bk_ = load_b_frag(KsU, lane, t * 16, 32, 16 * ks);
                sacc[t] = wmma_bf16(a, bk_, sacc[t]);
            }
        }

        // ---- scale + causal mask ----
        const int qg  = qb * 64 + 16 * wave + rl0;  // + r
        const int kg0 = kb * 64 + cl;               // + 16*t
        float sv[4][8];
#pragma unroll
        for (int t = 0; t < 4; ++t)
#pragma unroll
            for (int r = 0; r < 8; ++r) {
                const float v = sacc[t][r] * 0.125f;  // 1/sqrt(64)
                sv[t][r] = (kg0 + 16 * t <= qg + r) ? v : -1e30f;
            }

        // ---- online softmax update; write P (bf16) to LDS ----
#pragma unroll
        for (int r = 0; r < 8; ++r) {
            float mx = fmaxf(fmaxf(sv[0][r], sv[1][r]),
                             fmaxf(sv[2][r], sv[3][r]));
#pragma unroll
            for (int off = 1; off < 16; off <<= 1)
                mx = fmaxf(mx, __shfl_xor(mx, off, 32));
            const float mnew  = fmaxf(m_i[r], mx);
            const float alpha = __expf(m_i[r] - mnew);
            float rs = 0.0f;
#pragma unroll
            for (int t = 0; t < 4; ++t) {
                const float p = __expf(sv[t][r] - mnew);
                rs += p;
                Ps[(16 * wave + rl0 + r) * 64 + t * 16 + cl] = f2bf(p);
            }
#pragma unroll
            for (int off = 1; off < 16; off <<= 1)
                rs += __shfl_xor(rs, off, 32);
            l_i[r] = l_i[r] * alpha + rs;
            m_i[r] = mnew;
#pragma unroll
            for (int t = 0; t < 4; ++t) oacc[t][r] *= alpha;
        }
        // Each wave reads back only its own P rows; per-wave LDS ops are
        // in-order, so no barrier needed between the P store and the loads.

        // ---- O += P @ V  (8 WMMAs) ----
#pragma unroll
        for (int ks = 0; ks < 2; ++ks) {
            v16bf a = load_a_frag(PsU, lane, 16 * wave, 32, 16 * ks);
#pragma unroll
            for (int t = 0; t < 4; ++t) {
                v16bf bv = load_b_frag(VsU, lane, t * 16, 32, 16 * ks);
                oacc[t] = wmma_bf16(a, bv, oacc[t]);
            }
        }
    }

    // ---- normalize and store O (bf16) ----
#pragma unroll
    for (int r = 0; r < 8; ++r) {
        const float inv = 1.0f / l_i[r];
        const size_t row = rowQ0 + 16 * wave + rl0 + r;
#pragma unroll
        for (int t = 0; t < 4; ++t)
            out[row * H + h * HS + t * 16 + cl] = f2bf(oacc[t][r] * inv);
    }
}

// ---------------------------------------------------------------------------
extern "C" void kernel_launch(void* const* d_in, const int* in_sizes, int n_in,
                              void* d_out, int out_size, void* d_ws, size_t ws_size,
                              hipStream_t stream) {
    (void)in_sizes; (void)n_in; (void)out_size; (void)ws_size;

    const float* x      = (const float*)d_in[0];   // [8,1024,768]
    const float* W_attn = (const float*)d_in[1];   // [768,2304]
    const float* b_attn = (const float*)d_in[2];   // [2304]
    const float* W_proj = (const float*)d_in[3];   // [768,768]
    const float* b_proj = (const float*)d_in[4];   // [768]
    float* out = (float*)d_out;                    // [8,1024,768]

    constexpr int M = 8 * 1024;      // 8192 rows (B*T)
    constexpr int H = 768;
    constexpr int QKV = 3 * 768;     // 2304

    unsigned short* qkv_ws = (unsigned short*)d_ws;            // 37.75 MB bf16
    unsigned short* att_ws = qkv_ws + (size_t)M * QKV;         // 12.58 MB bf16

    // 1) qkv = x @ W_attn + b_attn   (f32 in, bf16 out)
    gemm_bias_kernel<false, true>
        <<<dim3(QKV / 64, M / 64), 256, 0, stream>>>(
            x, W_attn, b_attn, qkv_ws, M, QKV, H);

    // 2) flash attention per (batch, head, 64-row q block)
    attn_kernel<<<dim3(16, 96), 128, 0, stream>>>(qkv_ws, att_ws);

    // 3) out = attn @ W_proj + b_proj   (bf16 in, f32 out)
    gemm_bias_kernel<true, false>
        <<<dim3(H / 64, M / 64), 256, 0, stream>>>(
            att_ws, W_proj, b_proj, out, M, H, H);
}
